// self_attention_11622181503693
// MI455X (gfx1250) — matrix operand order repaired and checked by reading, not compiled, not run on hardware
//
#include <hip/hip_runtime.h>

typedef __attribute__((ext_vector_type(16))) __bf16 v16bf;
typedef __attribute__((ext_vector_type(8)))  __bf16 v8bf;
typedef __attribute__((ext_vector_type(8)))  float  v8f;

#define B_  16
#define H_  512
#define L_  2048
#define HL_ (H_*L_)

// ---------------- fragment helpers ----------------

// A-matrix 16x32 bf16 fragment from row-major LDS tile (rows x stride),
// contiguous k order (same as swizzle_b_kernel): lane n (n<16): M=n, elems 0-15 = K[k0..k0+15];
// lanes 16-31: +16 on the K base.
__device__ __forceinline__ v16bf ldA(const __bf16* base, int stride, int k0, int lane) {
  int m = lane & 15, hi = lane >> 4;
  const __bf16* p = base + m * stride + k0 + 16 * hi;
  v8bf lo = *(const v8bf*)p;
  v8bf hh = *(const v8bf*)(p + 8);
  return __builtin_shufflevector(lo, hh, 0,1,2,3,4,5,6,7,8,9,10,11,12,13,14,15);
}

// B-matrix 32x16 bf16 fragment from pre-swizzled tiles: tile = 32 lanes * 16 elems,
// lane-linear so each lane does one 32-byte contiguous load.
__device__ __forceinline__ v16bf ldB(const __bf16* tiles, long long tileIdx, int lane) {
  return *(const v16bf*)(tiles + tileIdx * 512 + lane * 16);
}

__device__ __forceinline__ v8f wmma_bf16(v16bf a, v16bf b, v8f c) {
  return __builtin_amdgcn_wmma_f32_16x16x32_bf16(false, a, false, b, (short)0, c, false, false);
}

// ---------------- prep: build swizzled bf16 B-operand tiles ----------------
// dst tile t holds B[k][n] with k = kt*32 + (lane>>4)*16 + e, n = nt*16 + (lane&15),
// element = src[b*batchStride + k*kStride + n*nStride].
__global__ void swizzle_b_kernel(const float* __restrict__ src, __bf16* __restrict__ dst,
                                 int Ktiles, int Ntiles,
                                 long long kStride, long long nStride,
                                 long long batchStride, int totalWaves) {
  int wid = blockIdx.x * (blockDim.x >> 5) + (threadIdx.x >> 5);
  if (wid >= totalWaves) return;
  int lane = threadIdx.x & 31;
  int tilesPerBatch = Ktiles * Ntiles;
  int b  = wid / tilesPerBatch;
  int r  = wid - b * tilesPerBatch;
  int kt = r / Ntiles;
  int nt = r - kt * Ntiles;
  const float* s = src + (long long)b * batchStride;
  long long k0 = (long long)kt * 32 + ((lane >> 4) << 4);
  long long n  = (long long)nt * 16 + (lane & 15);
  __bf16* d = dst + (long long)wid * 512 + lane * 16;
#pragma unroll
  for (int e = 0; e < 16; ++e)
    d[e] = (__bf16)s[(k0 + e) * kStride + n * nStride];
}

// ---------------- fused attention ----------------
__global__ __launch_bounds__(256) void attn_fused(
    const float* __restrict__ ctx, const float* __restrict__ b_in, const float* __restrict__ b_out,
    const __bf16* __restrict__ ctx_sw, const __bf16* __restrict__ ctxT_sw,
    const __bf16* __restrict__ WinT_sw, const __bf16* __restrict__ WoutT_sw,
    float* __restrict__ out)
{
  __shared__ __bf16 sA[16 * 520];   // ctx^T tile  [16 rows l][512 h], pitch 520
  __shared__ __bf16 sG[16 * 520];   // gamma tile
  __shared__ float  sS[16 * 132];   // scores sub-block [16][128], pitch 132
  __shared__ __bf16 sP[16 * 136];   // softmax weights bf16, pitch 136
  __shared__ __bf16 sC[16 * 520];   // c_t bf16
  __shared__ float  sMax[16], sSum[16], sRes[16];

  const int b   = blockIdx.x >> 7;
  const int l0  = (blockIdx.x & 127) << 4;
  const int tid = threadIdx.x;
  const int w   = tid >> 5;        // wave 0..7
  const int lane = tid & 31;
  const int hi  = lane >> 4;
  const int nl  = lane & 15;

  const float* ctxb = ctx + (long long)b * HL_;

  // ---- Step 0: load ctx^T tile (16 x 512) as bf16 ----
  for (int idx = tid; idx < 16 * 512; idx += 256) {
    int row = idx & 15;
    int h   = idx >> 4;
    sA[row * 520 + h] = (__bf16)ctxb[(long long)h * L_ + l0 + row];
  }
  __syncthreads();

  // ---- Step 1: gamma = ctx^T_tile * W_in^T + b_in ----
  for (int t = 0; t < 4; ++t) {
    int nt = w * 4 + t;
    v8f acc = {0.f,0.f,0.f,0.f,0.f,0.f,0.f,0.f};
#pragma unroll 4
    for (int kt = 0; kt < 16; ++kt) {
      v16bf a  = ldA(sA, 520, kt * 32, lane);
      v16bf bb = ldB(WinT_sw, (long long)kt * 32 + nt, lane);
      acc = wmma_bf16(a, bb, acc);
    }
    float bias = b_in[nt * 16 + nl];
#pragma unroll
    for (int v = 0; v < 8; ++v) {
      int row = v + 8 * hi;
      sG[row * 520 + nt * 16 + nl] = (__bf16)(acc[v] + bias);
    }
  }
  if (tid < 16) { sMax[tid] = -3.0e38f; sSum[tid] = 0.f; }
  __syncthreads();

  // ---- Step 2: flash loop over key tiles of 128 ----
  v8f accC[4];
#pragma unroll
  for (int t = 0; t < 4; ++t) accC[t] = (v8f){0.f,0.f,0.f,0.f,0.f,0.f,0.f,0.f};

  const float scl = 0.044194173824159216f;  // 1/sqrt(512)

  for (int ktk = 0; ktk < 16; ++ktk) {
    // (a) scores sub-tile: wave w -> key cols [ktk*128 + w*16, +16)
    {
      long long ntg = (long long)ktk * 8 + w;
      v8f acc = {0.f,0.f,0.f,0.f,0.f,0.f,0.f,0.f};
#pragma unroll 4
      for (int kk = 0; kk < 16; ++kk) {
        v16bf a  = ldA(sG, 520, kk * 32, lane);
        v16bf bb = ldB(ctx_sw, ((long long)b * 16 + kk) * 128 + ntg, lane);
        acc = wmma_bf16(a, bb, acc);
      }
#pragma unroll
      for (int v = 0; v < 8; ++v) {
        int row = v + 8 * hi;
        sS[row * 132 + w * 16 + nl] = acc[v] * scl;
      }
    }
    __syncthreads();

    // (b) online softmax stats, fully parallel:
    // 8 waves x 2 rows; 16 lanes per row; 8 key-cols per lane;
    // butterfly __shfl_xor (masks 1/2/4/8 stay within each 16-lane half).
    {
      int row = 2 * w + hi;
      int cb  = nl * 8;
      float s0[8];
#pragma unroll
      for (int j = 0; j < 8; ++j) s0[j] = sS[row * 132 + cb + j];
      float m = s0[0];
#pragma unroll
      for (int j = 1; j < 8; ++j) m = fmaxf(m, s0[j]);
#pragma unroll
      for (int off = 1; off < 16; off <<= 1)
        m = fmaxf(m, __shfl_xor(m, off, 32));
      float mo = sMax[row];
      float mn = fmaxf(mo, m);
      float rs = __expf(mo - mn);
      float sum = 0.f;
      v8bf pv;
#pragma unroll
      for (int j = 0; j < 8; ++j) {
        float p = __expf(s0[j] - mn);
        sum += p;
        pv[j] = (__bf16)p;
      }
      *(v8bf*)(&sP[row * 136 + cb]) = pv;   // one ds_store_b128
#pragma unroll
      for (int off = 1; off < 16; off <<= 1)
        sum += __shfl_xor(sum, off, 32);
      if (nl == 0) {
        sMax[row] = mn;
        sSum[row] = sSum[row] * rs + sum;
        sRes[row] = rs;
      }
    }
    __syncthreads();

    // (c) rescale accumulators, accumulate P @ ctx^T (wave owns 64 h-cols)
    float rsv[8];
#pragma unroll
    for (int v = 0; v < 8; ++v) rsv[v] = sRes[v + 8 * hi];
#pragma unroll
    for (int t = 0; t < 4; ++t)
#pragma unroll
      for (int v = 0; v < 8; ++v) accC[t][v] *= rsv[v];

    for (int t = 0; t < 4; ++t) {
      int nt = w * 4 + t;
#pragma unroll
      for (int kk = 0; kk < 4; ++kk) {
        v16bf a  = ldA(sP, 136, kk * 32, lane);
        v16bf bb = ldB(ctxT_sw, ((long long)b * 64 + ktk * 4 + kk) * 32 + nt, lane);
        accC[t] = wmma_bf16(a, bb, accC[t]);
      }
    }
    __syncthreads();
  }

  // ---- Step 3: normalize, write c_t as bf16 to LDS ----
  float invs[8];
#pragma unroll
  for (int v = 0; v < 8; ++v) invs[v] = 1.0f / sSum[v + 8 * hi];
#pragma unroll
  for (int t = 0; t < 4; ++t) {
    int nt = w * 4 + t;
#pragma unroll
    for (int v = 0; v < 8; ++v) {
      int row = v + 8 * hi;
      sC[row * 520 + nt * 16 + nl] = (__bf16)(accC[t][v] * invs[v]);
    }
  }
  __syncthreads();

  // ---- Step 4: out = [c_t | ctx^T] * W_out^T + b_out, SELU, transposed store ----
  for (int t = 0; t < 4; ++t) {
    int nt = w * 4 + t;
    v8f acc = {0.f,0.f,0.f,0.f,0.f,0.f,0.f,0.f};
#pragma unroll 4
    for (int kt = 0; kt < 16; ++kt) {
      v16bf a  = ldA(sC, 520, kt * 32, lane);
      v16bf bb = ldB(WoutT_sw, (long long)kt * 32 + nt, lane);
      acc = wmma_bf16(a, bb, acc);
    }
#pragma unroll 4
    for (int kt = 0; kt < 16; ++kt) {
      v16bf a  = ldA(sA, 520, kt * 32, lane);
      v16bf bb = ldB(WoutT_sw, (long long)(16 + kt) * 32 + nt, lane);
      acc = wmma_bf16(a, bb, acc);
    }
    int o = nt * 16 + nl;
    float bo = b_out[o];
    float vals[8];
#pragma unroll
    for (int v = 0; v < 8; ++v) {
      float x = acc[v] + bo;
      vals[v] = x > 0.f ? 1.0507009873554805f * x
                        : 1.7580993408473766f * (__expf(x) - 1.f);
    }
    float* op = out + (long long)b * HL_ + (long long)o * L_ + l0 + 8 * hi;
    *(float4*)op       = make_float4(vals[0], vals[1], vals[2], vals[3]);
    *(float4*)(op + 4) = make_float4(vals[4], vals[5], vals[6], vals[7]);
  }
}

// ---------------- host launch ----------------
extern "C" void kernel_launch(void* const* d_in, const int* in_sizes, int n_in,
                              void* d_out, int out_size, void* d_ws, size_t ws_size,
                              hipStream_t stream) {
  (void)in_sizes; (void)n_in; (void)out_size; (void)ws_size;
  const float* ctx   = (const float*)d_in[0];
  const float* W_in  = (const float*)d_in[1];
  const float* b_in  = (const float*)d_in[2];
  const float* W_out = (const float*)d_in[3];
  const float* b_out = (const float*)d_in[4];
  float* out = (float*)d_out;

  char* ws = (char*)d_ws;
  __bf16* ctx_sw   = (__bf16*)(ws);                                   // 32 MB: (b,16 kt,128 nt)
  __bf16* ctxT_sw  = (__bf16*)(ws + (size_t)32 * 1024 * 1024);        // 32 MB: (b,64 kt, 32 nt)
  __bf16* WinT_sw  = (__bf16*)(ws + (size_t)64 * 1024 * 1024);        // 512 KB: (16 kt, 32 nt)
  __bf16* WoutT_sw = (__bf16*)(ws + (size_t)64 * 1024 * 1024 + 512 * 1024); // 1 MB: (32 kt, 32 nt)

  // scores B operand: B[k=h][n=m] = ctx[b][h][m]
  { int waves = 16 * 16 * 128;
    swizzle_b_kernel<<<(waves + 7) / 8, 256, 0, stream>>>(ctx, ctx_sw, 16, 128,
        (long long)L_, 1LL, (long long)HL_, waves); }
  // PV B operand: B[k=m][n=h] = ctx[b][h][m]
  { int waves = 16 * 64 * 32;
    swizzle_b_kernel<<<(waves + 7) / 8, 256, 0, stream>>>(ctx, ctxT_sw, 64, 32,
        1LL, (long long)L_, (long long)HL_, waves); }
  // gamma B operand: B[k=h][n=o] = W_in[o][h]
  { int waves = 16 * 32;
    swizzle_b_kernel<<<(waves + 7) / 8, 256, 0, stream>>>(W_in, WinT_sw, 16, 32,
        1LL, (long long)H_, 0LL, waves); }
  // out B operand: B[k][n=o] = W_out[o][k], k in [0,1024)
  { int waves = 32 * 32;
    swizzle_b_kernel<<<(waves + 7) / 8, 256, 0, stream>>>(W_out, WoutT_sw, 32, 32,
        1LL, 1024LL, 0LL, waves); }

  attn_fused<<<B_ * (L_ / 16), 256, 0, stream>>>(ctx, b_in, b_out,
      ctx_sw, ctxT_sw, WinT_sw, WoutT_sw, out);
}
